// EnhancedGAT_5257039970571
// MI455X (gfx1250) — compile-verified, matrix operand order checked by reading, not run on hardware
//
#include <hip/hip_runtime.h>
#include <hip/hip_bf16.h>
#include <math.h>

typedef __attribute__((ext_vector_type(16))) _Float16 v16h;
typedef __attribute__((ext_vector_type(8)))  float    v8f;

#define GN      100000
#define GE      1600000
#define GEN     (GE + GN)
#define KDIM    128
#define GHID    128
#define GHEADS  4
#define GNC     40
#define NEGS    0.2f
#define BN_EPS  1e-5f

// ---------------------------------------------------------------------------
// Fragment loader: for lane half `hi`, the 16-bit A/B fragment for k-block kb
// is two contiguous 8-float runs of the row: [kb+8*hi, +8) and [kb+16+8*hi, +8).
// Load as 4x b128, convert f32->f16 in registers. Branch-free.
// ---------------------------------------------------------------------------
__device__ __forceinline__ v16h load_frag_f16(const float* __restrict__ row, int kb, int hi) {
    const float4 l0 = *(const float4*)(row + kb + 8 * hi);
    const float4 l1 = *(const float4*)(row + kb + 8 * hi + 4);
    const float4 h0 = *(const float4*)(row + kb + 16 + 8 * hi);
    const float4 h1 = *(const float4*)(row + kb + 16 + 8 * hi + 4);
    v16h f;
    f[0]  = (_Float16)l0.x; f[1]  = (_Float16)l0.y; f[2]  = (_Float16)l0.z; f[3]  = (_Float16)l0.w;
    f[4]  = (_Float16)l1.x; f[5]  = (_Float16)l1.y; f[6]  = (_Float16)l1.z; f[7]  = (_Float16)l1.w;
    f[8]  = (_Float16)h0.x; f[9]  = (_Float16)h0.y; f[10] = (_Float16)h0.z; f[11] = (_Float16)h0.w;
    f[12] = (_Float16)h1.x; f[13] = (_Float16)h1.y; f[14] = (_Float16)h1.z; f[15] = (_Float16)h1.w;
    return f;
}

// ---------------------------------------------------------------------------
// WMMA GEMM: P[n][j] = sum_k X[n][k] * W[j][k]   (X: Nrows x 128, W: Out x 128)
// One wave per 16x16 tile; K fully unrolled -> 4 chained v_wmma, f32 accum.
// Out-of-range rows/cols load from clamped addresses; their garbage stays in
// D rows/cols that the store guard discards (WMMA row/col independence).
// ---------------------------------------------------------------------------
__global__ void gat_gemm_wmma(const float* __restrict__ X, const float* __restrict__ W,
                              float* __restrict__ P, int Nrows, int Out, int ldP) {
    const int lane = threadIdx.x & 31;
    const int m    = lane & 15;
    const int hi   = lane >> 4;
    const int row0 = blockIdx.x * 16;
    const int col0 = blockIdx.y * 16;

    int rA = row0 + m; if (rA >= Nrows) rA = Nrows - 1;   // clamp: safe address
    int cB = col0 + m; if (cB >= Out)   cB = Out - 1;     // clamp: safe address
    const float* xr = X + (size_t)rA * KDIM;
    const float* wr = W + (size_t)cB * KDIM;

    v8f acc = {};
#pragma unroll
    for (int kb = 0; kb < KDIM; kb += 32) {
        const v16h a = load_frag_f16(xr, kb, hi);
        const v16h b = load_frag_f16(wr, kb, hi);
        acc = __builtin_amdgcn_wmma_f32_16x16x32_f16(false, a, false, b,
                                                     (short)0, acc, false, false);
    }
#pragma unroll
    for (int i = 0; i < 8; ++i) {
        const int row = row0 + i + 8 * hi;
        const int col = col0 + m;
        if (row < Nrows && col < Out) P[(size_t)row * ldP + col] = acc[i];
    }
}

// ---------------------------------------------------------------------------
// Per-node attention logits: als[n,h] = <h[n,h,:], a_s[h,:]>, same for ald.
// ---------------------------------------------------------------------------
__global__ void gat_att_logits(const float* __restrict__ P,
                               const float* __restrict__ a_s, const float* __restrict__ a_d,
                               float* __restrict__ als, float* __restrict__ ald,
                               int H, int C, int ld) {
    const int idx = blockIdx.x * blockDim.x + threadIdx.x;
    if (idx >= GN * H) return;
    const int n = idx / H, h = idx - n * H;
    const float* p = P + (size_t)n * ld + h * C;
    float s = 0.f, d = 0.f;
    for (int c = 0; c < C; ++c) {
        const float v = p[c];
        s = fmaf(v, a_s[h * C + c], s);
        d = fmaf(v, a_d[h * C + c], d);
    }
    als[idx] = s;
    ald[idx] = d;
}

__global__ void gat_fill(float* __restrict__ p, float v, size_t n) {
    const size_t i = (size_t)blockIdx.x * blockDim.x + threadIdx.x;
    if (i < n) p[i] = v;
}

__device__ __forceinline__ void atomicMaxFloat(float* addr, float val) {
    unsigned* ua = (unsigned*)addr;
    unsigned old = *ua;
    while (__uint_as_float(old) < val) {
        const unsigned assumed = old;
        old = atomicCAS(ua, assumed, __float_as_uint(val));
        if (old == assumed) break;
    }
}

// edge e >= GE is the implicit self loop (src = dst = e - GE)
__device__ __forceinline__ void edge_ends(const int* __restrict__ src,
                                          const int* __restrict__ dst,
                                          int e, int& s, int& d) {
    if (e < GE) { s = src[e]; d = dst[e]; }
    else        { s = d = e - GE; }
}

// ---------------------------------------------------------------------------
// Edge pass 1: e = leaky_relu(als[src]+ald[dst]); segment max over dst.
// ---------------------------------------------------------------------------
__global__ void gat_edge_logit_max(const int* __restrict__ src, const int* __restrict__ dst,
                                   const float* __restrict__ als, const float* __restrict__ ald,
                                   float* __restrict__ ew, float* __restrict__ mbuf, int H) {
    const int idx = blockIdx.x * blockDim.x + threadIdx.x;
    if (idx >= GEN * H) return;
    const int e = idx / H, h = idx - e * H;
    int s, d; edge_ends(src, dst, e, s, d);
    float v = als[s * H + h] + ald[d * H + h];
    v = v > 0.f ? v : NEGS * v;
    ew[idx] = v;
    atomicMaxFloat(&mbuf[d * H + h], v);
}

// ---------------------------------------------------------------------------
// Edge pass 2: ex = exp(e - m[dst]); segment sum over dst.
// ---------------------------------------------------------------------------
__global__ void gat_edge_exp_sum(const int* __restrict__ src, const int* __restrict__ dst,
                                 const float* __restrict__ mbuf, float* __restrict__ ew,
                                 float* __restrict__ ssum, int H) {
    const int idx = blockIdx.x * blockDim.x + threadIdx.x;
    if (idx >= GEN * H) return;
    const int e = idx / H, h = idx - e * H;
    int s, d; edge_ends(src, dst, e, s, d);
    const float x = __expf(ew[idx] - mbuf[d * H + h]);
    ew[idx] = x;
    atomicAdd(&ssum[d * H + h], x);
}

// ---------------------------------------------------------------------------
// Edge pass 3: out[dst] += alpha * h[src]. One wave32 per edge; lane = feature.
// ---------------------------------------------------------------------------
__global__ void gat_edge_scatter(const int* __restrict__ src, const int* __restrict__ dst,
                                 const float* __restrict__ ew, const float* __restrict__ ssum,
                                 const float* __restrict__ P, float* __restrict__ O,
                                 int H, int C, int ld) {
    const int e = blockIdx.x * (blockDim.x >> 5) + (threadIdx.x >> 5);
    if (e >= GEN) return;
    const int lane = threadIdx.x & 31;
    int s, d; edge_ends(src, dst, e, s, d);
    for (int f = lane; f < ld; f += 32) {
        const int h = f / C;
        const float alpha = ew[e * H + h] / ssum[d * H + h];
        atomicAdd(&O[(size_t)d * ld + f], alpha * P[(size_t)s * ld + f]);
    }
}

// ---------------------------------------------------------------------------
// Epilogue layers 0/1: +bias, BatchNorm(eval), ReLU.
// ---------------------------------------------------------------------------
__global__ void gat_bias_bn_relu(const float* __restrict__ O, const float* __restrict__ bias,
                                 const float* __restrict__ g, const float* __restrict__ be,
                                 const float* __restrict__ mn, const float* __restrict__ vr,
                                 float* __restrict__ A, int ld) {
    const int idx = blockIdx.x * blockDim.x + threadIdx.x;
    if (idx >= GN * ld) return;
    const int j = idx % ld;
    float v = O[idx] + bias[j];
    v = (v - mn[j]) * rsqrtf(vr[j] + BN_EPS) * g[j] + be[j];
    A[idx] = v > 0.f ? v : 0.f;
}

// ---------------------------------------------------------------------------
// Final epilogue: +bias, log_softmax over 40 classes. One wave32 per node;
// lane covers class c = lane and c = lane + 32, wave-reduce via shfl_xor.
// ---------------------------------------------------------------------------
__global__ void gat_bias_logsoftmax(const float* __restrict__ O, const float* __restrict__ bias,
                                    float* __restrict__ out, int C) {
    const int node = blockIdx.x * (blockDim.x >> 5) + (threadIdx.x >> 5);
    if (node >= GN) return;
    const int lane = threadIdx.x & 31;
    const float v0 = (lane < C) ? O[(size_t)node * C + lane] + bias[lane] : -INFINITY;
    const float v1 = (lane + 32 < C) ? O[(size_t)node * C + lane + 32] + bias[lane + 32] : -INFINITY;
    float mx = fmaxf(v0, v1);
#pragma unroll
    for (int off = 16; off > 0; off >>= 1) mx = fmaxf(mx, __shfl_xor(mx, off, 32));
    float s = (lane < C ? __expf(v0 - mx) : 0.f) + (lane + 32 < C ? __expf(v1 - mx) : 0.f);
#pragma unroll
    for (int off = 16; off > 0; off >>= 1) s += __shfl_xor(s, off, 32);
    const float ls = __logf(s);
    if (lane < C)      out[(size_t)node * C + lane]      = v0 - mx - ls;
    if (lane + 32 < C) out[(size_t)node * C + lane + 32] = v1 - mx - ls;
}

// ---------------------------------------------------------------------------
// Host orchestration
// ---------------------------------------------------------------------------
static void run_gat_layer(const float* in, const float* W, const float* a_s, const float* a_d,
                          int H, int C, int Out, float* P, float* O,
                          float* ALS, float* ALD, float* M, float* S, float* EW,
                          const int* src, const int* dst, hipStream_t stream) {
    const int ld = Out;
    // 1. projection (WMMA)
    gat_gemm_wmma<<<dim3((GN + 15) / 16, (Out + 15) / 16), 32, 0, stream>>>(in, W, P, GN, Out, ld);
    // 2. per-node logits
    gat_att_logits<<<(GN * H + 255) / 256, 256, 0, stream>>>(P, a_s, a_d, ALS, ALD, H, C, ld);
    // 3. init segment stats + accumulator
    gat_fill<<<(GN * H + 255) / 256, 256, 0, stream>>>(M, -INFINITY, (size_t)GN * H);
    gat_fill<<<(GN * H + 255) / 256, 256, 0, stream>>>(S, 0.f, (size_t)GN * H);
    gat_fill<<<((size_t)GN * ld + 255) / 256, 256, 0, stream>>>(O, 0.f, (size_t)GN * ld);
    // 4-6. edge passes
    gat_edge_logit_max<<<(GEN * H + 255) / 256, 256, 0, stream>>>(src, dst, ALS, ALD, EW, M, H);
    gat_edge_exp_sum<<<(GEN * H + 255) / 256, 256, 0, stream>>>(src, dst, M, EW, S, H);
    gat_edge_scatter<<<(GEN + 7) / 8, 256, 0, stream>>>(src, dst, EW, S, P, O, H, C, ld);
}

extern "C" void kernel_launch(void* const* d_in, const int* in_sizes, int n_in,
                              void* d_out, int out_size, void* d_ws, size_t ws_size,
                              hipStream_t stream) {
    const float* x   = (const float*)d_in[0];
    const int*   ei  = (const int*)  d_in[1];
    const float* W0  = (const float*)d_in[2];
    const float* as0 = (const float*)d_in[3];
    const float* ad0 = (const float*)d_in[4];
    const float* b0  = (const float*)d_in[5];
    const float* g0  = (const float*)d_in[6];
    const float* be0 = (const float*)d_in[7];
    const float* m0  = (const float*)d_in[8];
    const float* v0  = (const float*)d_in[9];
    const float* W1  = (const float*)d_in[10];
    const float* as1 = (const float*)d_in[11];
    const float* ad1 = (const float*)d_in[12];
    const float* b1  = (const float*)d_in[13];
    const float* g1  = (const float*)d_in[14];
    const float* be1 = (const float*)d_in[15];
    const float* m1  = (const float*)d_in[16];
    const float* v1  = (const float*)d_in[17];
    const float* W2  = (const float*)d_in[18];
    const float* as2 = (const float*)d_in[19];
    const float* ad2 = (const float*)d_in[20];
    const float* b2  = (const float*)d_in[21];

    const int* src = ei;        // edge_index[0]
    const int* dst = ei + GE;   // edge_index[1]

    float* ws  = (float*)d_ws;
    float* P   = ws;                          // [GN,128] projection
    float* O   = P + (size_t)GN * GHID;       // [GN,128] scatter accumulator
    float* A   = O + (size_t)GN * GHID;       // [GN,128] layer activation
    float* ALS = A + (size_t)GN * GHID;       // [GN,4]
    float* ALD = ALS + (size_t)GN * GHEADS;   // [GN,4]
    float* M   = ALD + (size_t)GN * GHEADS;   // [GN,4] segment max
    float* S   = M + (size_t)GN * GHEADS;     // [GN,4] segment sum
    float* EW  = S + (size_t)GN * GHEADS;     // [GEN,4] edge weights

    // ---- layer 0: x(128) -> 4 heads x 32, concat ----
    run_gat_layer(x, W0, as0, ad0, GHEADS, GHID / GHEADS, GHID,
                  P, O, ALS, ALD, M, S, EW, src, dst, stream);
    gat_bias_bn_relu<<<((size_t)GN * GHID + 255) / 256, 256, 0, stream>>>(O, b0, g0, be0, m0, v0, A, GHID);

    // ---- layer 1: A(128) -> 4 heads x 32, concat ----
    run_gat_layer(A, W1, as1, ad1, GHEADS, GHID / GHEADS, GHID,
                  P, O, ALS, ALD, M, S, EW, src, dst, stream);
    gat_bias_bn_relu<<<((size_t)GN * GHID + 255) / 256, 256, 0, stream>>>(O, b1, g1, be1, m1, v1, A, GHID);

    // ---- layer 2: A(128) -> 1 head x 40, no concat ----
    run_gat_layer(A, W2, as2, ad2, 1, GNC, GNC,
                  P, O, ALS, ALD, M, S, EW, src, dst, stream);
    gat_bias_logsoftmax<<<(GN + 7) / 8, 256, 0, stream>>>(O, b2, (float*)d_out, GNC);
}
